// MaskedSelfAttention_26826365731383
// MI455X (gfx1250) — compile-verified, matrix operand order checked by reading, not run on hardware
//
#include <hip/hip_runtime.h>
#include <hip/hip_bf16.h>

#define B_ 4
#define S_ 2048
#define E_ 1024
#define QT_ (S_/16)   // 128 query tiles per batch

typedef __bf16 bf16_t;
typedef __attribute__((ext_vector_type(16))) __bf16 v16bf;
typedef __attribute__((ext_vector_type(8)))  float  v8f;

// ---------------------------------------------------------------------------
// Fragment loader for 16x32 bf16 tiles (A operand row-major, or B operand as
// B^T row-major). Per ISA 16-bit A layout: lanes 0-15 -> rows 0-15 with
// K {0..7} in VGPR0-3 and K {16..23} in VGPR4-7; lanes 16-31 -> same rows,
// K {8..15} and {24..31}. Two 16-byte loads per lane. Works for both global
// and LDS source pointers (generic addressing).
// ---------------------------------------------------------------------------
__device__ __forceinline__ v16bf load_frag16x32(const bf16_t* base, int ld, int lane) {
  const int lr = lane & 15;
  const int hf = (lane >> 4) & 1;
  const bf16_t* p = base + (long)lr * ld + hf * 8;
  union { uint4 u[2]; v16bf v; } r;
  r.u[0] = *(const uint4*)(p);
  r.u[1] = *(const uint4*)(p + 16);
  return r.v;
}

// ---------------------------------------------------------------------------
// Kernel 0: fp32 -> bf16 conversion, 8 elements per thread
// ---------------------------------------------------------------------------
__global__ void cvt_f32_bf16(const float* __restrict__ src,
                             bf16_t* __restrict__ dst, long n8) {
  long i = (long)blockIdx.x * blockDim.x + threadIdx.x;
  if (i >= n8) return;
  const float4* s = (const float4*)src;
  float4 a = s[2 * i], b = s[2 * i + 1];
  union { bf16_t h[8]; uint4 u; } pk;
  pk.h[0] = (bf16_t)a.x; pk.h[1] = (bf16_t)a.y;
  pk.h[2] = (bf16_t)a.z; pk.h[3] = (bf16_t)a.w;
  pk.h[4] = (bf16_t)b.x; pk.h[5] = (bf16_t)b.y;
  pk.h[6] = (bf16_t)b.z; pk.h[7] = (bf16_t)b.w;
  ((uint4*)dst)[i] = pk.u;
}

// ---------------------------------------------------------------------------
// Kernel 1: QKV projection, one matrix per blockIdx.y, 64 rows per block.
// x tile (64x1024 bf16, 128 KB) staged in LDS once; wave w owns features
// [w*128, w*128+128) and keeps 4 Mtile x 8 Ntile f32 accumulators (256 VGPRs)
// so each weight B-fragment is reused 4x from registers.
// q,k stored bf16 row-major [B*S][E]; v stored transposed [B][E][S].
// ---------------------------------------------------------------------------
__global__ __launch_bounds__(256) void qkv_kernel(
    const bf16_t* __restrict__ xb,
    const bf16_t* __restrict__ wq, const bf16_t* __restrict__ wk,
    const bf16_t* __restrict__ wv,
    const float* __restrict__ bq, const float* __restrict__ bk,
    const float* __restrict__ bv,
    bf16_t* __restrict__ qb, bf16_t* __restrict__ kb,
    bf16_t* __restrict__ vtb) {
  extern __shared__ char smem_raw[];
  bf16_t* xs = (bf16_t*)smem_raw;            // 64 x 1024 bf16 = 128 KB

  const int mat = blockIdx.y;                // 0=q, 1=k, 2=v
  const bf16_t* W   = (mat == 0) ? wq : (mat == 1) ? wk : wv;
  const float* bias = (mat == 0) ? bq : (mat == 1) ? bk : bv;

  const int tid  = threadIdx.x;
  const int w    = tid >> 5;
  const int lane = tid & 31;
  const int lr   = lane & 15;
  const int hf   = (lane >> 4) & 1;
  const long m0  = (long)blockIdx.x * 64;    // row in flattened [B*S, E]
  const int  fs  = w * 128;

  // Cooperative stage of the 64-row x tile into LDS (8192 x 16B).
  {
    const uint4* src = (const uint4*)(xb + m0 * E_);
    uint4* dst = (uint4*)xs;
    for (int i = tid; i < (64 * E_ * 2) / 16; i += 256) dst[i] = src[i];
  }
  __syncthreads();

  v8f acc[4][8] = {};
  for (int e0 = 0; e0 < E_; e0 += 32) {
    v16bf A[4];
#pragma unroll
    for (int mt = 0; mt < 4; ++mt)
      A[mt] = load_frag16x32(xs + (mt * 16) * E_ + e0, E_, lane);
#pragma unroll
    for (int nt = 0; nt < 8; ++nt) {
      v16bf Bf = load_frag16x32(W + (long)(fs + nt * 16) * E_ + e0, E_, lane);
#pragma unroll
      for (int mt = 0; mt < 4; ++mt)
        acc[mt][nt] = __builtin_amdgcn_wmma_f32_16x16x32_bf16(
            false, A[mt], false, Bf, (short)0, acc[mt][nt], false, false);
    }
  }

  const int bidx = (int)(m0 / S_);
  const int s0   = (int)(m0 % S_);
  if (mat < 2) {
    bf16_t* outp = (mat == 0) ? qb : kb;
#pragma unroll
    for (int nt = 0; nt < 8; ++nt) {
      const int f = fs + nt * 16 + lr;
      const float bvv = bias[f];
#pragma unroll
      for (int mt = 0; mt < 4; ++mt)
#pragma unroll
        for (int r = 0; r < 8; ++r) {
          const long row = m0 + mt * 16 + r + 8 * hf;  // D: M=r+8*hf, N=lr
          outp[row * E_ + f] = (bf16_t)(acc[mt][nt][r] + bvv);
        }
    }
  } else {
#pragma unroll
    for (int nt = 0; nt < 8; ++nt) {
      const int f = fs + nt * 16 + lr;
      const float bvv = bias[f];
#pragma unroll
      for (int mt = 0; mt < 4; ++mt) {
        union { bf16_t h[8]; uint4 u; } pk;
#pragma unroll
        for (int r = 0; r < 8; ++r) pk.h[r] = (bf16_t)(acc[mt][nt][r] + bvv);
        // s = s0 + mt*16 + 8*hf + r : 8 contiguous bf16 -> one 16B store
        *(uint4*)(vtb + ((long)bidx * E_ + f) * S_ + s0 + mt * 16 + 8 * hf) = pk.u;
      }
    }
  }
}

// ---------------------------------------------------------------------------
// Kernel 2a: scores = q k^T / 32, causal mask, softmax -> P (bf16).
// grid = (S/32, B): each block covers TWO query tiles (32 rows). Waves 0-3
// own tile 0, waves 4-7 own tile 1; each wave hoists its q A-fragments into
// 256 VGPRs. Each k tile is read once per 32 query rows. LDS: 32x2048 fp32
// scores (256 KB dynamic). P zero-padded to the pair extent (qp+1)*32 keys.
// ---------------------------------------------------------------------------
__global__ __launch_bounds__(256) void attn_scores_kernel(
    const bf16_t* __restrict__ qb, const bf16_t* __restrict__ kb,
    bf16_t* __restrict__ P) {
  extern __shared__ float sc[];  // 32 * S_ floats
  const int tid  = threadIdx.x;
  const int w    = tid >> 5;
  const int lane = tid & 31;
  const int lr   = lane & 15;
  const int hf   = (lane >> 4) & 1;
  const int b    = blockIdx.y;
  const int qp   = blockIdx.x;         // pair index, 0..S_/32-1
  const int t    = w >> 2;             // which query tile of the pair
  const int kg   = w & 3;              // key-tile round-robin group
  const int qt   = 2 * qp + t;
  const int q0   = qt * 16;

  v16bf afrag[32];
  const bf16_t* qbase = qb + (long)(b * S_ + q0) * E_;
#pragma unroll
  for (int i = 0; i < 32; ++i)
    afrag[i] = load_frag16x32(qbase + i * 32, E_, lane);

  for (int kt = kg; kt <= qt; kt += 4) {
    v8f c = {};
    const bf16_t* kbase = kb + (long)(b * S_ + kt * 16) * E_;
#pragma unroll
    for (int i = 0; i < 32; ++i) {
      v16bf Bf = load_frag16x32(kbase + i * 32, E_, lane);
      c = __builtin_amdgcn_wmma_f32_16x16x32_bf16(false, afrag[i], false, Bf,
                                                  (short)0, c, false, false);
    }
#pragma unroll
    for (int r = 0; r < 8; ++r) {
      const int M   = r + 8 * hf;
      const int col = kt * 16 + lr;
      float v = c[r] * 0.03125f;               // 1/sqrt(1024)
      if (kt == qt && lr > M) v = -3.0e30f;    // causal mask, diagonal tile
      sc[(t * 16 + M) * S_ + col] = v;
    }
  }
  __syncthreads();

  // Softmax: 8 threads per row over 32 rows; shuffle reductions width 8.
  const int row = tid >> 3;            // 0..31 within the pair
  const int l8  = tid & 7;
  const int rqt = 2 * qp + (row >> 4);
  const int L     = (rqt + 1) * 16;
  const int Lpair = (qp + 1) * 32;     // multiple of 32, >= L for both tiles
  float m = -3.4e38f;
  for (int col = l8; col < L; col += 8) m = fmaxf(m, sc[row * S_ + col]);
#pragma unroll
  for (int off = 4; off >= 1; off >>= 1) m = fmaxf(m, __shfl_xor(m, off, 8));
  float s = 0.f;
  for (int col = l8; col < L; col += 8) {
    const float e = __expf(sc[row * S_ + col] - m);
    sc[row * S_ + col] = e;
    s += e;
  }
#pragma unroll
  for (int off = 4; off >= 1; off >>= 1) s += __shfl_xor(s, off, 8);
  const float inv = 1.f / s;
  bf16_t* Prow = P + ((long)(b * QT_ + rqt) * 16 + (row & 15)) * S_;
  for (int col = l8; col < Lpair; col += 8) {
    const float p = (col < L) ? sc[row * S_ + col] * inv : 0.f;
    Prow[col] = (bf16_t)p;
  }
}

// ---------------------------------------------------------------------------
// Kernel 2b: attn = P * V, + residual x, LayerNorm -> out (fp32).
// grid = (S/32, B): 32 rows (two query tiles) per block; each v^T B-fragment
// feeds two WMMAs. LDS: 32x1024 fp32 y tile (128 KB dynamic).
// ---------------------------------------------------------------------------
__global__ __launch_bounds__(256) void attn_out_kernel(
    const bf16_t* __restrict__ P, const bf16_t* __restrict__ vtb,
    const float* __restrict__ x, const float* __restrict__ gamma,
    const float* __restrict__ beta, float* __restrict__ out) {
  extern __shared__ float yb[];  // 32 * E_ floats
  const int tid  = threadIdx.x;
  const int w    = tid >> 5;
  const int lane = tid & 31;
  const int lr   = lane & 15;
  const int hf   = (lane >> 4) & 1;
  const int b    = blockIdx.y;
  const int qp   = blockIdx.x;
  const int q0   = qp * 32;
  const int Lpair = (qp + 1) * 32;
  const int fs   = w * 128;

  v8f acc[2][8] = {};
  const bf16_t* P0 = P + (long)(b * QT_ + 2 * qp) * 16 * S_;  // pair rows contiguous
  for (int kk = 0; kk < Lpair; kk += 32) {
    v16bf A0 = load_frag16x32(P0 + kk, S_, lane);
    v16bf A1 = load_frag16x32(P0 + (long)16 * S_ + kk, S_, lane);
#pragma unroll
    for (int nt = 0; nt < 8; ++nt) {
      const bf16_t* vbp = vtb + ((long)b * E_ + fs + nt * 16) * S_ + kk;
      v16bf Bf = load_frag16x32(vbp, S_, lane);
      acc[0][nt] = __builtin_amdgcn_wmma_f32_16x16x32_bf16(
          false, A0, false, Bf, (short)0, acc[0][nt], false, false);
      acc[1][nt] = __builtin_amdgcn_wmma_f32_16x16x32_bf16(
          false, A1, false, Bf, (short)0, acc[1][nt], false, false);
    }
  }
#pragma unroll
  for (int nt = 0; nt < 8; ++nt) {
    const int f = fs + nt * 16 + lr;
#pragma unroll
    for (int t = 0; t < 2; ++t)
#pragma unroll
      for (int r = 0; r < 8; ++r)
        yb[(t * 16 + r + 8 * hf) * E_ + f] = acc[t][nt][r];
  }
  __syncthreads();

  // Residual + LayerNorm: 8 threads per row, 128 cols each, y updated
  // in place in LDS (each slot owned by exactly one thread).
  const int row = tid >> 3;
  const int l8  = tid & 7;
  const float* xr = x + ((long)(b * S_) + q0 + row) * E_;
  float sum = 0.f;
  for (int j = 0; j < 128; ++j) {
    const int c = l8 + j * 8;
    const float y = yb[row * E_ + c] + xr[c];
    yb[row * E_ + c] = y;
    sum += y;
  }
#pragma unroll
  for (int off = 4; off >= 1; off >>= 1) sum += __shfl_xor(sum, off, 8);
  const float mu = sum * (1.0f / E_);
  float vs = 0.f;
  for (int j = 0; j < 128; ++j) {
    const float d = yb[row * E_ + l8 + j * 8] - mu;
    vs += d * d;
  }
#pragma unroll
  for (int off = 4; off >= 1; off >>= 1) vs += __shfl_xor(vs, off, 8);
  const float rstd = rsqrtf(vs * (1.0f / E_) + 1e-5f);
  float* orow = out + ((long)(b * S_) + q0 + row) * E_;
  for (int j = 0; j < 128; ++j) {
    const int c = l8 + j * 8;
    orow[c] = (yb[row * E_ + c] - mu) * rstd * gamma[c] + beta[c];
  }
}

// ---------------------------------------------------------------------------
extern "C" void kernel_launch(void* const* d_in, const int* in_sizes, int n_in,
                              void* d_out, int out_size, void* d_ws, size_t ws_size,
                              hipStream_t stream) {
  (void)in_sizes; (void)n_in; (void)out_size; (void)ws_size;
  const float* x     = (const float*)d_in[0];
  // d_in[1] is the boolean causal mask; causality is applied analytically.
  const float* Wq    = (const float*)d_in[2];
  const float* bq    = (const float*)d_in[3];
  const float* Wk    = (const float*)d_in[4];
  const float* bk    = (const float*)d_in[5];
  const float* Wv    = (const float*)d_in[6];
  const float* bv    = (const float*)d_in[7];
  const float* gamma = (const float*)d_in[8];
  const float* beta  = (const float*)d_in[9];

  char* ws = (char*)d_ws;
  size_t off = 0;
  bf16_t* xb  = (bf16_t*)(ws + off); off += (size_t)B_ * S_ * E_ * 2;  // 16 MB
  bf16_t* wqb = (bf16_t*)(ws + off); off += (size_t)E_ * E_ * 2;       //  2 MB
  bf16_t* wkb = (bf16_t*)(ws + off); off += (size_t)E_ * E_ * 2;
  bf16_t* wvb = (bf16_t*)(ws + off); off += (size_t)E_ * E_ * 2;
  bf16_t* qb  = (bf16_t*)(ws + off); off += (size_t)B_ * S_ * E_ * 2;  // 16 MB
  bf16_t* kb  = (bf16_t*)(ws + off); off += (size_t)B_ * S_ * E_ * 2;  // 16 MB
  bf16_t* vtb = (bf16_t*)(ws + off); off += (size_t)B_ * E_ * S_ * 2;  // 16 MB
  bf16_t* Pw  = (bf16_t*)(ws + off); off += (size_t)B_ * S_ * S_ * 2;  // 32 MB

  const long n8x = (long)B_ * S_ * E_ / 8;
  const long n8w = (long)E_ * E_ / 8;
  cvt_f32_bf16<<<(n8x + 255) / 256, 256, 0, stream>>>(x, xb, n8x);
  cvt_f32_bf16<<<(n8w + 255) / 256, 256, 0, stream>>>(Wq, wqb, n8w);
  cvt_f32_bf16<<<(n8w + 255) / 256, 256, 0, stream>>>(Wk, wkb, n8w);
  cvt_f32_bf16<<<(n8w + 255) / 256, 256, 0, stream>>>(Wv, wvb, n8w);

  const int qkvLds = 64 * E_ * 2;          // 128 KB
  hipFuncSetAttribute(reinterpret_cast<const void*>(qkv_kernel),
                      hipFuncAttributeMaxDynamicSharedMemorySize, qkvLds);
  qkv_kernel<<<dim3(B_ * S_ / 64, 3), 256, qkvLds, stream>>>(
      xb, wqb, wkb, wvb, bq, bk, bv, qb, kb, vtb);

  const int scLds = 32 * S_ * (int)sizeof(float);   // 256 KB
  hipFuncSetAttribute(reinterpret_cast<const void*>(attn_scores_kernel),
                      hipFuncAttributeMaxDynamicSharedMemorySize, scLds);
  attn_scores_kernel<<<dim3(S_ / 32, B_), 256, scLds, stream>>>(qb, kb, Pw);

  const int outLds = 32 * E_ * (int)sizeof(float);  // 128 KB
  hipFuncSetAttribute(reinterpret_cast<const void*>(attn_out_kernel),
                      hipFuncAttributeMaxDynamicSharedMemorySize, outLds);
  attn_out_kernel<<<dim3(S_ / 32, B_), 256, outLds, stream>>>(
      Pw, vtb, x, gamma, beta, (float*)d_out);
}
